// SEGNN_44203803411116
// MI455X (gfx1250) — matrix operand order repaired and checked by reading, not compiled, not run on hardware
//
#include <hip/hip_runtime.h>
#include <hip/hip_bf16.h>
#include <math.h>

// ---------------------------------------------------------------------------
// SEGNN forward for MI455X (gfx1250, wave32).
// TP-linear layers run as f16 WMMA (v_wmma_f32_16x16x32_f16) with f32
// accumulation.  Each wave owns 16 rows (edges/nodes).  GEMMs are split per
// tensor-product region; packed B buffers carry a K-permutation so that A
// fragments are built in registers with no divisions and minimal LDS traffic
// (outer-product structure: one y-load per K-chunk, x preloaded per lane).
// ---------------------------------------------------------------------------

#define NNODES 20000
#define NEDGES 320000

typedef _Float16 f16t;
typedef _Float16 v16h __attribute__((ext_vector_type(16)));
typedef float    v8f  __attribute__((ext_vector_type(8)));
typedef unsigned int v8u __attribute__((ext_vector_type(8)));

#define RSQ3 0.57735026918962576f
#define SQ3  1.7320508075688772f

struct Bset  { const f16t* sa; const f16t* sb; const f16t* va; const f16t* vb; };
struct Bset3 { Bset l1, l2, l3; };

__device__ __forceinline__ v8f wmma_f32_f16(v16h a, v16h b, v8f c) {
  return __builtin_amdgcn_wmma_f32_16x16x32_f16(false, a, false, b, (short)0, c, false, false);
}

// B fragment (32x16 f16): pre-packed so each lane's 16 halves are contiguous.
__device__ __forceinline__ v16h load_b_frag(const f16t* B, int kt, int nt, int NT, int lane) {
  const uint4* p = (const uint4*)(B + ((size_t)((kt * NT + nt) * 32 + lane) << 4));
  uint4 x = p[0];
  uint4 y = p[1];
  v8u t;
  t[0] = x.x; t[1] = x.y; t[2] = x.z; t[3] = x.w;
  t[4] = y.x; t[5] = y.y; t[6] = y.z; t[7] = y.w;
  return __builtin_bit_cast(v16h, t);
}

__device__ __forceinline__ float gelu_tanh(float x) {
  float x3 = x * x * x;
  return 0.5f * x * (1.0f + tanhf(0.7978845608028654f * (x + 0.044715f * x3)));
}
__device__ __forceinline__ float sigmoid_(float x) { return 1.0f / (1.0f + expf(-x)); }

template<int NT>
__device__ __forceinline__ void zacc(v8f (&acc)[NT]) {
#pragma unroll
  for (int nt = 0; nt < NT; ++nt)
#pragma unroll
    for (int rr = 0; rr < 8; ++rr) acc[nt][rr] = 0.0f;
}

// A[r,k] = Xr[k*xstep]*scale, k < Kvalid (branchless clamp+select).
template<int NT>
__device__ __forceinline__ void gemm_direct(v8f (&acc)[NT], const f16t* B, int KT, int Kvalid,
                                            const float* Xr, int xstep, float scale, int lane) {
  int kb0 = (lane & 16) ? 8 : 0;
  for (int kt = 0; kt < KT; ++kt) {
    int k0 = (kt << 5) + kb0;
    v16h a;
#pragma unroll
    for (int t = 0; t < 8; ++t) {
      int ka = k0 + t, kb = k0 + 16 + t;
      int kac = (ka < Kvalid) ? ka : (Kvalid - 1);
      int kbc = (kb < Kvalid) ? kb : (Kvalid - 1);
      float va = Xr[kac * xstep] * scale;
      float vb = Xr[kbc * xstep] * scale;
      a[t]     = (f16t)((ka < Kvalid) ? va : 0.0f);
      a[8 + t] = (f16t)((kb < Kvalid) ? vb : 0.0f);
    }
#pragma unroll
    for (int nt = 0; nt < NT; ++nt)
      acc[nt] = wmma_f32_f16(a, load_b_frag(B, kt, nt, NT, lane), acc[nt]);
  }
}

// A[r,k] = dot3(x, Y[k]) * RSQ3, k < Kvalid.
template<int NT>
__device__ __forceinline__ void gemm_dot3_y(v8f (&acc)[NT], const f16t* B, int KT, int Kvalid,
                                            float x0, float x1, float x2,
                                            const float* Yb, int lane) {
  int kb0 = (lane & 16) ? 8 : 0;
  for (int kt = 0; kt < KT; ++kt) {
    int k0 = (kt << 5) + kb0;
    v16h a;
#pragma unroll
    for (int t = 0; t < 8; ++t) {
      int ka = k0 + t, kb = k0 + 16 + t;
      int kac = (ka < Kvalid) ? ka : 0;
      int kbc = (kb < Kvalid) ? kb : 0;
      float va = (x0 * Yb[kac * 3] + x1 * Yb[kac * 3 + 1] + x2 * Yb[kac * 3 + 2]) * RSQ3;
      float vb = (x0 * Yb[kbc * 3] + x1 * Yb[kbc * 3 + 1] + x2 * Yb[kbc * 3 + 2]) * RSQ3;
      a[t]     = (f16t)((ka < Kvalid) ? va : 0.0f);
      a[8 + t] = (f16t)((kb < Kvalid) ? vb : 0.0f);
    }
#pragma unroll
    for (int nt = 0; nt < NT; ++nt)
      acc[nt] = wmma_f32_f16(a, load_b_frag(B, kt, nt, NT, lane), acc[nt]);
  }
}

// K-permuted outer product, k' = j*32 + i (i<32 exact): A[r,k'] = X[i]*Y[j].
// Per chunk: j = kt constant -> one y-load; x preloaded (16 regs).
template<int NT>
__device__ __forceinline__ void gemm_swap32(v8f (&acc)[NT], const f16t* B, int KT,
                                            const float* Xr, const float* Yr, int ystep, int lane) {
  int kb0 = (lane & 16) ? 8 : 0;
  float x0[8], x1[8];
#pragma unroll
  for (int t = 0; t < 8; ++t) { x0[t] = Xr[kb0 + t]; x1[t] = Xr[kb0 + 16 + t]; }
  for (int kt = 0; kt < KT; ++kt) {
    float yv = Yr[kt * ystep];
    if (kt + 1 < KT)
      __builtin_prefetch(B + ((size_t)((kt + 1) * NT * 32 + lane) << 4), 0, 1);
    v16h a;
#pragma unroll
    for (int t = 0; t < 8; ++t) { a[t] = (f16t)(x0[t] * yv); a[8 + t] = (f16t)(x1[t] * yv); }
#pragma unroll
    for (int nt = 0; nt < NT; ++nt)
      acc[nt] = wmma_f32_f16(a, load_b_frag(B, kt, nt, NT, lane), acc[nt]);
  }
}

// K-permuted, k' = j*16 + i (i padded to 16): A[r,k'] = dot3(X[i],Y[j])*RSQ3.
// Per chunk j = {2kt, 2kt+1}; x rows preloaded (RSQ3 pre-scaled, zero pad).
template<int NT>
__device__ __forceinline__ void gemm_swap16_dot3(v8f (&acc)[NT], const f16t* B, int KT, int jmax,
                                                 const float* Xb, int imax, const float* Yb,
                                                 int lane) {
  int kb0 = (lane & 16) ? 8 : 0;
  float xr0[8], xr1[8], xr2[8];
#pragma unroll
  for (int t = 0; t < 8; ++t) {
    int i = kb0 + t;
    int ic = (i < imax) ? i : 0;
    float g = (i < imax) ? RSQ3 : 0.0f;
    xr0[t] = Xb[ic * 3 + 0] * g;
    xr1[t] = Xb[ic * 3 + 1] * g;
    xr2[t] = Xb[ic * 3 + 2] * g;
  }
  for (int kt = 0; kt < KT; ++kt) {
    int j0 = 2 * kt, j1 = 2 * kt + 1;
    int j0c = (j0 < jmax) ? j0 : 0;
    int j1c = (j1 < jmax) ? j1 : 0;
    float g0 = (j0 < jmax) ? 1.0f : 0.0f;
    float g1 = (j1 < jmax) ? 1.0f : 0.0f;
    float ya0 = Yb[j0c * 3 + 0] * g0, ya1 = Yb[j0c * 3 + 1] * g0, ya2 = Yb[j0c * 3 + 2] * g0;
    float yb0 = Yb[j1c * 3 + 0] * g1, yb1 = Yb[j1c * 3 + 1] * g1, yb2 = Yb[j1c * 3 + 2] * g1;
    v16h a;
#pragma unroll
    for (int t = 0; t < 8; ++t) {
      a[t]     = (f16t)(xr0[t] * ya0 + xr1[t] * ya1 + xr2[t] * ya2);
      a[8 + t] = (f16t)(xr0[t] * yb0 + xr1[t] * yb1 + xr2[t] * yb2);
    }
#pragma unroll
    for (int nt = 0; nt < NT; ++nt)
      acc[nt] = wmma_f32_f16(a, load_b_frag(B, kt, nt, NT, lane), acc[nt]);
  }
}

// K-permuted, k' = j*16 + i: A[r,k'] = X[i]*Y[j].
template<int NT>
__device__ __forceinline__ void gemm_swap16_mul(v8f (&acc)[NT], const f16t* B, int KT, int jmax,
                                                const float* Xb, int xstep, int imax,
                                                const float* Yb, int ystep, int lane) {
  int kb0 = (lane & 16) ? 8 : 0;
  float xr[8];
#pragma unroll
  for (int t = 0; t < 8; ++t) {
    int i = kb0 + t;
    int ic = (i < imax) ? i : 0;
    xr[t] = (i < imax) ? Xb[ic * xstep] : 0.0f;
  }
  for (int kt = 0; kt < KT; ++kt) {
    int j0 = 2 * kt, j1 = 2 * kt + 1;
    int j0c = (j0 < jmax) ? j0 : 0;
    int j1c = (j1 < jmax) ? j1 : 0;
    float y0 = (j0 < jmax) ? Yb[j0c * ystep] : 0.0f;
    float y1 = (j1 < jmax) ? Yb[j1c * ystep] : 0.0f;
    v16h a;
#pragma unroll
    for (int t = 0; t < 8; ++t) { a[t] = (f16t)(xr[t] * y0); a[8 + t] = (f16t)(xr[t] * y1); }
#pragma unroll
    for (int nt = 0; nt < NT; ++nt)
      acc[nt] = wmma_f32_f16(a, load_b_frag(B, kt, nt, NT, lane), acc[nt]);
  }
}

// D layout: lane L -> col = L%16; VGPR r -> row = r + (L>=16 ? 8 : 0).
template<int NT>
__device__ __forceinline__ void store_s(float* db, v8f (&acc)[NT], int lane) {
  int col = lane & 15, mb = (lane >> 4) << 3;
#pragma unroll
  for (int nt = 0; nt < NT; ++nt)
#pragma unroll
    for (int rr = 0; rr < 8; ++rr)
      db[(mb + rr) * 48 + nt * 16 + col] = acc[nt][rr];
}
__device__ __forceinline__ void store_v(float* db, v8f (&acc)[1], int d, int lane) {
  int col = lane & 15, mb = (lane >> 4) << 3;
  if (col < 10)
#pragma unroll
    for (int rr = 0; rr < 8; ++rr)
      db[((mb + rr) * 10 + col) * 3 + d] = acc[0][rr];
}

// ---------------------------------------------------------------------------
// Edge geometry: a_v = sqrt(3)*rhat; cnt = in-degree; aiv = sum(a_v) by recv.
// ---------------------------------------------------------------------------
__global__ __launch_bounds__(256) void geom_kernel(
    const float* __restrict__ pos, const int* __restrict__ snd,
    const int* __restrict__ rcv, float* __restrict__ av,
    float* __restrict__ cnt, float* __restrict__ aiv, int E) {
  int e = blockIdx.x * 256 + threadIdx.x;
  if (e >= E) return;
  int s = snd[e], r = rcv[e];
  float dx = pos[s * 3 + 0] - pos[r * 3 + 0];
  float dy = pos[s * 3 + 1] - pos[r * 3 + 1];
  float dz = pos[s * 3 + 2] - pos[r * 3 + 2];
  float n = sqrtf(dx * dx + dy * dy + dz * dz) + 1e-8f;
  float ax = SQ3 * dx / n, ay = SQ3 * dy / n, az = SQ3 * dz / n;
  av[e * 3 + 0] = ax; av[e * 3 + 1] = ay; av[e * 3 + 2] = az;
  atomicAdd(&cnt[r], 1.0f);
  atomicAdd(&aiv[r * 3 + 0], ax);
  atomicAdd(&aiv[r * 3 + 1], ay);
  atomicAdd(&aiv[r * 3 + 2], az);
}

// ---------------------------------------------------------------------------
// Pack fp32 weight W[rows][Ncols] into fragment-ordered f16 with an optional
// K-permutation: swap -> orig row = rowBase + (k&mamask)*sb + (k>>mashift),
// valid iff (k&mamask)<imax && (k>>mashift)<jmax.  identity -> row = rowBase+k,
// valid iff k < jmax.
// ---------------------------------------------------------------------------
__global__ __launch_bounds__(128) void pack_b_kernel(
    const float* __restrict__ W, f16t* __restrict__ dst, int rowBase, int swap,
    int mamask, int mashift, int sb_, int imax, int jmax, int Ncols, int NT, int total) {
  int t = blockIdx.x * 128 + threadIdx.x;
  if (t >= total) return;
  int lane = t & 31;
  int tile = t >> 5;
  int kt = tile / NT, nt = tile % NT;
  int col = nt * 16 + (lane & 15);
  int kbase = kt * 32 + ((lane & 16) ? 8 : 0);
  f16t h[16];
#pragma unroll
  for (int i = 0; i < 16; ++i) {
    int k = kbase + i + ((i >= 8) ? 8 : 0);
    int row;
    bool valid;
    if (swap) {
      int ii = k & mamask, jj = k >> mashift;
      valid = (ii < imax) && (jj < jmax);
      row = rowBase + ii * sb_ + jj;
    } else {
      valid = k < jmax;
      row = rowBase + k;
    }
    valid = valid && (col < Ncols);
    row = valid ? row : 0;
    int cc = valid ? col : 0;
    float v = W[row * Ncols + cc];
    h[i] = (f16t)(valid ? v : 0.0f);
  }
  uint4* d = (uint4*)(dst + (size_t)t * 16);
  d[0] = *(const uint4*)&h[0];
  d[1] = *(const uint4*)&h[8];
}

// Edge layers 2/3: x = (32 scal, 10 vec), y = (1, a_v).
template<int NTS>
__device__ __forceinline__ void edge_layer23(const Bset& L, float* douts_b, float* doutv_b,
                                             const float* hs_r, const float* hv_r,
                                             float e0, float e1, float e2, int lane) {
  v8f accS[NTS]; zacc(accS);
  gemm_direct<NTS>(accS, L.sa, 1, 32, hs_r, 1, 1.0f, lane);
  gemm_dot3_y<NTS>(accS, L.sb, 1, 10, e0, e1, e2, hv_r, lane);
  store_s<NTS>(douts_b, accS, lane);
  for (int d = 0; d < 3; ++d) {
    float ed = (d == 0) ? e0 : ((d == 1) ? e1 : e2);
    v8f aV[1]; zacc(aV);
    gemm_direct<1>(aV, L.va, 1, 32, hs_r, 1, ed, lane);
    gemm_direct<1>(aV, L.vb, 1, 10, hv_r + d, 3, 1.0f, lane);
    store_v(doutv_b, aV, d, lane);
  }
}

// Node layer, x = (32 scal, NVX vec), y = (33, 11).
template<int NTS>
__device__ __forceinline__ void node_layer_std(const Bset& L, float* douts_b, float* doutv_b,
                                               const float* Xs_r, const float* Xv_r,
                                               const float* ys_r, const float* yv_r,
                                               int NVX, int lane) {
  v8f accS[NTS]; zacc(accS);
  gemm_swap32<NTS>(accS, L.sa, 33, Xs_r, ys_r, 1, lane);           // ss: 32x33
  gemm_swap16_dot3<NTS>(accS, L.sb, 6, 11, Xv_r, NVX, yv_r, lane); // vv: NVXx11
  store_s<NTS>(douts_b, accS, lane);
  for (int d = 0; d < 3; ++d) {
    v8f aV[1]; zacc(aV);
    gemm_swap32<1>(aV, L.va, 11, Xs_r, yv_r + d, 3, lane);                   // sv: 32x11
    gemm_swap16_mul<1>(aV, L.vb, 17, 33, Xv_r + d, 3, NVX, ys_r, 1, lane);   // vs: NVXx33
    store_v(doutv_b, aV, d, lane);
  }
}

// ---------------------------------------------------------------------------
// Edge phase: gather -> 3 TP-linear layers (WMMA) -> atomic scatter by recv.
// 64 threads = 2 waves, 16 edges/wave.
// ---------------------------------------------------------------------------
template<int NSIN, int NVIN>
__global__ __launch_bounds__(64) void edge_kernel(
    const float* __restrict__ ns_in, const float* __restrict__ nv_in,
    const float* __restrict__ av, const int* __restrict__ snd,
    const int* __restrict__ rcv, Bset3 Bx,
    const float* __restrict__ bs1, const float* __restrict__ bs2,
    const float* __restrict__ bs3,
    float* __restrict__ sum_s, float* __restrict__ sum_v, int E) {
  constexpr int GS = (2 * NSIN > 0) ? 2 * NSIN : 1;
  constexpr int GV = 2 * NVIN;

  __shared__ float gxs[2][16][GS];
  __shared__ float gxv[2][16][GV][3];
  __shared__ float eav[2][16][3];
  __shared__ int   sidx[2][16], ridx[2][16];
  __shared__ float hs[2][16][32];
  __shared__ float hv[2][16][10][3];
  __shared__ float douts[2][16][48];
  __shared__ float doutv[2][16][10][3];

  const int lane = threadIdx.x & 31;
  const int w = threadIdx.x >> 5;
  const int e0i = (blockIdx.x * 2 + w) * 16;

  if (lane < 16) {
    int e = e0i + lane; if (e >= E) e = E - 1;
    sidx[w][lane] = snd[e];
    ridx[w][lane] = rcv[e];
  }
  __builtin_amdgcn_wave_barrier();
  for (int idx = lane; idx < 48; idx += 32) {
    int m = idx / 3, d = idx % 3;
    int e = e0i + m; if (e >= E) e = E - 1;
    eav[w][m][d] = av[e * 3 + d];
  }
  if (NSIN > 0) {
    for (int idx = lane; idx < 16 * GS; idx += 32) {
      int m = idx / GS, j = idx % GS;
      int node = (j < NSIN) ? sidx[w][m] : ridx[w][m];
      int jj = (j < NSIN) ? j : j - NSIN;
      gxs[w][m][j] = ns_in[node * NSIN + jj];
    }
  }
  for (int idx = lane; idx < 16 * GV * 3; idx += 32) {
    int m = idx / (GV * 3), rr = idx % (GV * 3), j = rr / 3, d = rr % 3;
    int node = (j < NVIN) ? sidx[w][m] : ridx[w][m];
    int jj = (j < NVIN) ? j : j - NVIN;
    gxv[w][m][j][d] = nv_in[(node * NVIN + jj) * 3 + d];
  }
  __builtin_amdgcn_wave_barrier();

  const int r = lane & 15;
  const float e0 = eav[w][r][0], e1 = eav[w][r][1], e2 = eav[w][r][2];
  const float* gxs_r = &gxs[w][r][0];
  const float* gxv_r = &gxv[w][r][0][0];
  const float* hs_r  = &hs[w][r][0];
  const float* hv_r  = &hv[w][r][0][0];
  float* douts_b = &douts[w][0][0];
  float* doutv_b = &doutv[w][0][0][0];

  // ----- layer 1: x = (2*NSIN scal, 2*NVIN vec), y = (1, a_v) -----
  {
    v8f accS[3]; zacc(accS);
    if constexpr (NSIN > 0)
      gemm_direct<3>(accS, Bx.l1.sa, (2 * NSIN) / 32, 2 * NSIN, gxs_r, 1, 1.0f, lane);
    gemm_dot3_y<3>(accS, Bx.l1.sb, 1, 2 * NVIN, e0, e1, e2, gxv_r, lane);
    store_s<3>(douts_b, accS, lane);
    for (int d = 0; d < 3; ++d) {
      float ed = (d == 0) ? e0 : ((d == 1) ? e1 : e2);
      v8f aV[1]; zacc(aV);
      if constexpr (NSIN > 0)
        gemm_direct<1>(aV, Bx.l1.va, (2 * NSIN) / 32, 2 * NSIN, gxs_r, 1, ed, lane);
      gemm_direct<1>(aV, Bx.l1.vb, 1, 2 * NVIN, gxv_r + d, 3, 1.0f, lane);
      store_v(doutv_b, aV, d, lane);
    }
  }
  __builtin_amdgcn_wave_barrier();
  for (int idx = lane; idx < 512; idx += 32) {
    int m = idx >> 5, i = idx & 31;
    hs[w][m][i] = gelu_tanh(douts[w][m][10 + i] + bs1[10 + i]);
  }
  for (int idx = lane; idx < 480; idx += 32) {
    int m = idx / 30, rr = idx % 30, q = rr / 3, d = rr % 3;
    hv[w][m][q][d] = doutv[w][m][q][d] * sigmoid_(douts[w][m][q] + bs1[q]);
  }
  __builtin_amdgcn_wave_barrier();

  edge_layer23<3>(Bx.l2, douts_b, doutv_b, hs_r, hv_r, e0, e1, e2, lane);
  __builtin_amdgcn_wave_barrier();
  for (int idx = lane; idx < 512; idx += 32) {
    int m = idx >> 5, i = idx & 31;
    hs[w][m][i] = gelu_tanh(douts[w][m][10 + i] + bs2[10 + i]);
  }
  for (int idx = lane; idx < 480; idx += 32) {
    int m = idx / 30, rr = idx % 30, q = rr / 3, d = rr % 3;
    hv[w][m][q][d] = doutv[w][m][q][d] * sigmoid_(douts[w][m][q] + bs2[q]);
  }
  __builtin_amdgcn_wave_barrier();

  edge_layer23<2>(Bx.l3, douts_b, doutv_b, hs_r, hv_r, e0, e1, e2, lane);
  __builtin_amdgcn_wave_barrier();

  // ----- scatter to receivers -----
  for (int idx = lane; idx < 512; idx += 32) {
    int m = idx >> 5, i = idx & 31;
    if (e0i + m < E)
      atomicAdd(&sum_s[ridx[w][m] * 32 + i], douts[w][m][i] + bs3[i]);
  }
  for (int idx = lane; idx < 480; idx += 32) {
    int m = idx / 30, rr = idx % 30, q = rr / 3, d = rr % 3;
    if (e0i + m < E)
      atomicAdd(&sum_v[(ridx[w][m] * 10 + q) * 3 + d], doutv[w][m][q][d]);
  }
}

// ---------------------------------------------------------------------------
// Node phase: aggregate -> 3 TP-linear layers (WMMA) -> residual -> write.
// ---------------------------------------------------------------------------
template<int NSIN, int NVIN>
__global__ __launch_bounds__(64) void node_kernel(
    const float* __restrict__ ns_in, const float* __restrict__ nv_in,
    const float* __restrict__ sum_s, const float* __restrict__ sum_v,
    const float* __restrict__ cnt, const float* __restrict__ aiv, Bset3 Bn,
    const float* __restrict__ bs1, const float* __restrict__ bs2,
    const float* __restrict__ bs3,
    const float* __restrict__ resWs, const float* __restrict__ resWv,
    float* __restrict__ ns_out, float* __restrict__ nv_out,
    float* __restrict__ fout, int N, float inv) {
  constexpr int XS = (NSIN > 0) ? NSIN : 1;

  __shared__ float xs[2][16][XS];
  __shared__ float xv[2][16][NVIN][3];
  __shared__ float ys[2][16][33];
  __shared__ float yv[2][16][11][3];
  __shared__ float hs[2][16][32];
  __shared__ float hv[2][16][10][3];
  __shared__ float douts[2][16][48];
  __shared__ float doutv[2][16][10][3];

  const int lane = threadIdx.x & 31;
  const int w = threadIdx.x >> 5;
  const int n0 = (blockIdx.x * 2 + w) * 16;

  if (NSIN > 0) {
    for (int idx = lane; idx < 16 * NSIN; idx += 32) {
      int m = idx / NSIN, i = idx % NSIN;
      int n = n0 + m; if (n >= N) n = N - 1;
      xs[w][m][i] = ns_in[n * NSIN + i];
    }
  }
  for (int idx = lane; idx < 16 * NVIN * 3; idx += 32) {
    int m = idx / (NVIN * 3), rr = idx % (NVIN * 3), j = rr / 3, d = rr % 3;
    int n = n0 + m; if (n >= N) n = N - 1;
    xv[w][m][j][d] = nv_in[(n * NVIN + j) * 3 + d];
  }
  for (int idx = lane; idx < 16 * 33; idx += 32) {
    int m = idx / 33, j = idx % 33;
    int n = n0 + m; if (n >= N) n = N - 1;
    float c = fmaxf(cnt[n], 1.0f);
    ys[w][m][j] = (j < 32) ? sum_s[n * 32 + j] / c * inv : cnt[n] * inv;
  }
  for (int idx = lane; idx < 16 * 33; idx += 32) {
    int m = idx / 33, rr = idx % 33, q = rr / 3, d = rr % 3;
    int n = n0 + m; if (n >= N) n = N - 1;
    float c = fmaxf(cnt[n], 1.0f);
    yv[w][m][q][d] = (q < 10) ? sum_v[(n * 10 + q) * 3 + d] / c * inv
                              : aiv[n * 3 + d] * inv;
  }
  __builtin_amdgcn_wave_barrier();

  const int r = lane & 15;
  const float* xs_r = &xs[w][r][0];
  const float* xv_r = &xv[w][r][0][0];
  const float* ys_r = &ys[w][r][0];
  const float* yv_r = &yv[w][r][0][0];
  const float* hs_r = &hs[w][r][0];
  const float* hv_r = &hv[w][r][0][0];
  float* douts_b = &douts[w][0][0];
  float* doutv_b = &doutv[w][0][0][0];

  // ----- layer 1 -----
  if constexpr (NSIN == 0) {
    float px0 = xv[w][r][0][0], px1 = xv[w][r][0][1], px2 = xv[w][r][0][2];
    v8f accS[3]; zacc(accS);
    gemm_dot3_y<3>(accS, Bn.l1.sb, 1, 11, px0, px1, px2, yv_r, lane);  // vv: 1x11
    store_s<3>(douts_b, accS, lane);
    for (int d = 0; d < 3; ++d) {
      float pxd = (d == 0) ? px0 : ((d == 1) ? px1 : px2);
      v8f aV[1]; zacc(aV);
      gemm_direct<1>(aV, Bn.l1.vb, 2, 33, ys_r, 1, pxd, lane);         // vs: 1x33
      store_v(doutv_b, aV, d, lane);
    }
  } else {
    node_layer_std<3>(Bn.l1, douts_b, doutv_b, xs_r, xv_r, ys_r, yv_r, NVIN, lane);
  }
  __builtin_amdgcn_wave_barrier();
  for (int idx = lane; idx < 512; idx += 32) {
    int m = idx >> 5, i = idx & 31;
    hs[w][m][i] = gelu_tanh(douts[w][m][10 + i] + bs1[10 + i]);
  }
  for (int idx = lane; idx < 480; idx += 32) {
    int m = idx / 30, rr = idx % 30, q = rr / 3, d = rr % 3;
    hv[w][m][q][d] = doutv[w][m][q][d] * sigmoid_(douts[w][m][q] + bs1[q]);
  }
  __builtin_amdgcn_wave_barrier();

  node_layer_std<3>(Bn.l2, douts_b, doutv_b, hs_r, hv_r, ys_r, yv_r, 10, lane);
  __builtin_amdgcn_wave_barrier();
  for (int idx = lane; idx < 512; idx += 32) {
    int m = idx >> 5, i = idx & 31;
    hs[w][m][i] = gelu_tanh(douts[w][m][10 + i] + bs2[10 + i]);
  }
  for (int idx = lane; idx < 480; idx += 32) {
    int m = idx / 30, rr = idx % 30, q = rr / 3, d = rr % 3;
    hv[w][m][q][d] = doutv[w][m][q][d] * sigmoid_(douts[w][m][q] + bs2[q]);
  }
  __builtin_amdgcn_wave_barrier();

  node_layer_std<2>(Bn.l3, douts_b, doutv_b, hs_r, hv_r, ys_r, yv_r, 10, lane);
  __builtin_amdgcn_wave_barrier();

  // ----- residual + write-out -----
  for (int idx = lane; idx < 512; idx += 32) {
    int m = idx >> 5, i = idx & 31;
    int n = n0 + m;
    if (n < N) {
      float rs = 0.0f;
      for (int p = 0; p < NSIN; ++p) rs += xs[w][m][p] * resWs[p * 32 + i];
      float v = douts[w][m][i] + bs3[i] + rs;
      ns_out[n * 32 + i] = v;
      if (fout) fout[n * 62 + i] = v;
    }
  }
  for (int idx = lane; idx < 480; idx += 32) {
    int m = idx / 30, rr = idx % 30, q = rr / 3, d = rr % 3;
    int n = n0 + m;
    if (n < N) {
      float rv = 0.0f;
      for (int p = 0; p < NVIN; ++p) rv += xv[w][m][p][d] * resWv[p * 10 + q];
      float v = doutv[w][m][q][d] + rv;
      nv_out[(n * 10 + q) * 3 + d] = v;
      if (fout) fout[n * 62 + 32 + q * 3 + d] = v;
    }
  }
}

// ---------------------------------------------------------------------------
// Host orchestration
// ---------------------------------------------------------------------------
extern "C" void kernel_launch(void* const* d_in, const int* in_sizes, int n_in,
                              void* d_out, int out_size, void* d_ws, size_t ws_size,
                              hipStream_t stream) {
  (void)in_sizes; (void)n_in; (void)out_size; (void)ws_size;
  const int N = NNODES, E = NEDGES;
  const float* pos = (const float*)d_in[0];
  const int* snd = (const int*)d_in[1];
  const int* rcv = (const int*)d_in[2];
  // jax pytree order: 3 steps x {edge[3]{Ws,Wv,bs}, node[3]{Ws,Wv,bs}, res{Ws,Wv}}
  auto leaf = [&](int i) -> const float* { return (const float*)d_in[3 + i]; };

  char* ws = (char*)d_ws;
  size_t off = 0;
  auto walloc = [&](size_t bytes) -> void* {
    void* p = ws + off;
    off = (off + bytes + 255) & ~(size_t)255;
    return p;
  };
  float* av    = (float*)walloc((size_t)E * 3 * 4);
  float* cnt   = (float*)walloc((size_t)N * 4);
  float* aiv   = (float*)walloc((size_t)N * 3 * 4);
  float* sum_s = (float*)walloc((size_t)N * 32 * 4);
  float* sum_v = (float*)walloc((size_t)N * 30 * 4);
  float* nsBuf[2] = {(float*)walloc((size_t)N * 32 * 4), (float*)walloc((size_t)N * 32 * 4)};
  float* nvBuf[2] = {(float*)walloc((size_t)N * 30 * 4), (float*)walloc((size_t)N * 30 * 4)};

  auto packId = [&](const float* W, int rowBase, int Kvalid, int Ncols, int KT, int NT) -> f16t* {
    int total = KT * NT * 32;
    f16t* dst = (f16t*)walloc((size_t)total * 16 * sizeof(f16t));
    pack_b_kernel<<<(total + 127) / 128, 128, 0, stream>>>(
        W, dst, rowBase, 0, 0, 0, 0, 0, Kvalid, Ncols, NT, total);
    return dst;
  };
  auto packSw = [&](const float* W, int rowBase, int mamask, int mashift, int sb_,
                    int imax, int jmax, int Ncols, int KT, int NT) -> f16t* {
    int total = KT * NT * 32;
    f16t* dst = (f16t*)walloc((size_t)total * 16 * sizeof(f16t));
    pack_b_kernel<<<(total + 127) / 128, 128, 0, stream>>>(
        W, dst, rowBase, 1, mamask, mashift, sb_, imax, jmax, Ncols, NT, total);
    return dst;
  };

  Bset3 eB[3], nB[3];
  for (int s = 0; s < 3; ++s) {
    int nsin = (s == 0) ? 0 : 32, nvin = (s == 0) ? 1 : 10;
    int base = s * 20;
    for (int b = 0; b < 3; ++b) {
      const float* Ws = leaf(base + b * 3 + 0);
      const float* Wv = leaf(base + b * 3 + 1);
      int R1 = (b == 0) ? 2 * nsin : 32;
      int R2 = (b == 0) ? 2 * nvin : 10;
      int Nc = (b == 2) ? 32 : 42;
      int NTs = (b == 2) ? 2 : 3;
      Bset L;
      L.sa = R1 ? packId(Ws, 0, R1, Nc, R1 / 32, NTs) : nullptr;
      L.sb = packId(Ws, R1, R2, Nc, 1, NTs);
      L.va = R1 ? packId(Wv, 0, R1, 10, R1 / 32, 1) : nullptr;
      L.vb = packId(Wv, R1, R2, 10, 1, 1);
      if (b == 0) eB[s].l1 = L; else if (b == 1) eB[s].l2 = L; else eB[s].l3 = L;
    }
    for (int b = 0; b < 3; ++b) {
      const float* Ws = leaf(base + 9 + b * 3 + 0);
      const float* Wv = leaf(base + 9 + b * 3 + 1);
      int nsx = (b == 0) ? nsin : 32, nvx = (b == 0) ? nvin : 10;
      int Nc = (b == 2) ? 32 : 42;
      int NTs = (b == 2) ? 2 : 3;
      Bset L;
      if (nsx == 0) {
        L.sa = nullptr;
        L.sb = packId(Ws, 0, nvx * 11, Nc, (nvx * 11 + 31) / 32, NTs);
        L.va = nullptr;
        L.vb = packId(Wv, 0, nvx * 33, 10, (nvx * 33 + 31) / 32, 1);
      } else {
        L.sa = packSw(Ws, 0, 31, 5, 33, 32, 33, Nc, 33, NTs);           // ss permuted
        L.sb = packSw(Ws, nsx * 33, 15, 4, 11, nvx, 11, Nc, 6, NTs);    // vv permuted
        L.va = packSw(Wv, 0, 31, 5, 11, 32, 11, 10, 11, 1);             // sv permuted
        L.vb = packSw(Wv, nsx * 11, 15, 4, 33, nvx, 33, 10, 17, 1);     // vs permuted
      }
      if (b == 0) nB[s].l1 = L; else if (b == 1) nB[s].l2 = L; else nB[s].l3 = L;
    }
  }

  hipMemsetAsync(cnt, 0, (size_t)N * 4, stream);
  hipMemsetAsync(aiv, 0, (size_t)N * 12, stream);
  geom_kernel<<<(E + 255) / 256, 256, 0, stream>>>(pos, snd, rcv, av, cnt, aiv, E);

  const float inv = 1.0f / (float)(E - 1);
  const float* nsIn = pos;  // unused when NSIN==0
  const float* nvIn = pos;  // step0: nv = pos as a single vector channel
  int cur = 0;
  for (int s = 0; s < 3; ++s) {
    hipMemsetAsync(sum_s, 0, (size_t)N * 32 * 4, stream);
    hipMemsetAsync(sum_v, 0, (size_t)N * 30 * 4, stream);
    int base = s * 20;
    const float *ebs1 = leaf(base + 2), *ebs2 = leaf(base + 5), *ebs3 = leaf(base + 8);
    const float *nbs1 = leaf(base + 11), *nbs2 = leaf(base + 14), *nbs3 = leaf(base + 17);
    const float *rWs = leaf(base + 18), *rWv = leaf(base + 19);
    float* nsO = nsBuf[cur];
    float* nvO = nvBuf[cur];
    float* fout = (s == 2) ? (float*)d_out : nullptr;
    int egrid = (E + 31) / 32, ngrid = (N + 31) / 32;
    if (s == 0) {
      edge_kernel<0, 1><<<egrid, 64, 0, stream>>>(
          nsIn, nvIn, av, snd, rcv, eB[s], ebs1, ebs2, ebs3, sum_s, sum_v, E);
      node_kernel<0, 1><<<ngrid, 64, 0, stream>>>(
          nsIn, nvIn, sum_s, sum_v, cnt, aiv, nB[s], nbs1, nbs2, nbs3, rWs, rWv,
          nsO, nvO, fout, N, inv);
    } else {
      edge_kernel<32, 10><<<egrid, 64, 0, stream>>>(
          nsIn, nvIn, av, snd, rcv, eB[s], ebs1, ebs2, ebs3, sum_s, sum_v, E);
      node_kernel<32, 10><<<ngrid, 64, 0, stream>>>(
          nsIn, nvIn, sum_s, sum_v, cnt, aiv, nB[s], nbs1, nbs2, nbs3, rWs, rWv,
          nsO, nvO, fout, N, inv);
    }
    nsIn = nsO;
    nvIn = nvO;
    cur ^= 1;
  }
}
